// AudioVisualModel_20066087207603
// MI455X (gfx1250) — matrix operand order repaired and checked by reading, not compiled, not run on hardware
//
#include <hip/hip_runtime.h>
#include <hip/hip_bf16.h>
#include <math.h>

typedef __bf16 bf16_t;
typedef __attribute__((ext_vector_type(16))) __bf16 v16bf;
typedef __attribute__((ext_vector_type(8)))  float   v8f;

#define BSZ    24
#define NA     50
#define NA_PAD 64
#define TT     8
#define NV     196
#define DD     768
#define NCOL   1568          /* TT*NV */
#define KSTEPS 24            /* DD/32 */
#define SSTR   1572          /* padded LDS row stride in floats (bank-conflict-free) */
#define NROWS  51            /* 50 real rows + 1 dump row for pad-row stores */
#define RED_OFF (NROWS * SSTR)
#define SMEM_FLOATS (RED_OFF + 512)
#define SMEM_BYTES  (SMEM_FLOATS * 4)   /* 322,736 B <= 327,680 B (320 KB) */

// ---------------------------------------------------------------------------
// Zero the global float accumulators (nonneg sum, temporal sum).
// ---------------------------------------------------------------------------
__global__ void init_kernel(float* wsf) {
    if (threadIdx.x < 2) wsf[threadIdx.x] = 0.0f;
}

// ---------------------------------------------------------------------------
// Normalize audio rows (L2 over D=768), convert to bf16, store row-major
// with contiguous K in each 32-K group (same order as V),
// so lanes<16 / lanes>=16 each read one
// contiguous 32B chunk per K-step. Rows 50..63 of each batch are zero pad.
// One wave per row; 8 rows per 256-thread block.
// ---------------------------------------------------------------------------
__global__ void norm_audio_kernel(const float* __restrict__ in, bf16_t* __restrict__ out) {
    const int row  = blockIdx.x * 8 + (threadIdx.x >> 5);   // 0 .. 24*64-1
    const int lane = threadIdx.x & 31;
    const int x = row >> 6;
    const int a = row & 63;

    // destination lane-slot within each 32-group (contiguous K, matches B)
    int dl = lane;

    bf16_t* orow = out + (size_t)row * DD;
    if (a >= NA) {
        #pragma unroll
        for (int j = 0; j < KSTEPS; ++j) orow[32 * j + dl] = (bf16_t)0.0f;
        return;
    }
    const float* irow = in + ((size_t)x * NA + a) * DD;
    float buf[KSTEPS];
    float ss = 0.0f;
    #pragma unroll
    for (int j = 0; j < KSTEPS; ++j) {
        float v = irow[32 * j + lane];
        buf[j] = v;
        ss += v * v;
    }
    #pragma unroll
    for (int o = 16; o > 0; o >>= 1) ss += __shfl_xor(ss, o, 32);
    const float inv = 1.0f / fmaxf(sqrtf(ss), 1e-12f);
    #pragma unroll
    for (int j = 0; j < KSTEPS; ++j) orow[32 * j + dl] = (bf16_t)(buf[j] * inv);
}

// ---------------------------------------------------------------------------
// Normalize visual rows; plain row-major bf16 (B-fragment reads contiguous K, same order as A).
// ---------------------------------------------------------------------------
__global__ void norm_visual_kernel(const float* __restrict__ in, bf16_t* __restrict__ out) {
    const int row  = blockIdx.x * 8 + (threadIdx.x >> 5);   // 0 .. 24*1568-1
    const int lane = threadIdx.x & 31;
    const float* irow = in + (size_t)row * DD;
    bf16_t*      orow = out + (size_t)row * DD;
    float buf[KSTEPS];
    float ss = 0.0f;
    #pragma unroll
    for (int j = 0; j < KSTEPS; ++j) {
        float v = irow[32 * j + lane];
        buf[j] = v;
        ss += v * v;
    }
    #pragma unroll
    for (int o = 16; o > 0; o >>= 1) ss += __shfl_xor(ss, o, 32);
    const float inv = 1.0f / fmaxf(sqrtf(ss), 1e-12f);
    #pragma unroll
    for (int j = 0; j < KSTEPS; ++j) orow[32 * j + lane] = (bf16_t)(buf[j] * inv);
}

// ---------------------------------------------------------------------------
// One workgroup per (x,y) batch pair.
// Phase 1: WMMA GEMM  S = A_x (50x768) * V_y^T (768x1568)  -> LDS (f32).
//          Pad rows (>=50) are clamped to a dump row: branch-free stores.
// Phase 2: all reductions fused out of LDS:
//   - max over Nv, threshold mask, temporal mean -> clip_sims[x,y]
//   - sum clip(s,-20,0)^2          -> atomicAdd wsf[0]
//   - sum (s[t]-s[t-1])^2 over nv  -> atomicAdd wsf[1]
// ---------------------------------------------------------------------------
__global__ __launch_bounds__(256) void gemm_pair_kernel(
    const bf16_t* __restrict__ A, const bf16_t* __restrict__ V,
    const float* __restrict__ tempP, const float* __restrict__ thrP,
    float* __restrict__ clip, float* __restrict__ wsf)
{
    extern __shared__ float smem[];
    float* S   = smem;            // NROWS x SSTR (row 50 = dump)
    float* red = smem + RED_OFF;  // [0,400): per-(a,t) max; [400,456): tok; [456,472): wave partials

    const int x = blockIdx.x, y = blockIdx.y;
    const int tid  = threadIdx.x;
    const int lane = tid & 31;
    const int wid  = tid >> 5;
    const int lm   = lane & 15;
    const int off  = (lane & 16) ? 16 : 0;   // element offset within each 32-K group

    const float invT = 1.0f / tempP[0];
    const float thr  = 1.0f / (1.0f + expf(-thrP[0]));

    const bf16_t* Ax = A + (size_t)x * NA_PAD * DD;
    const bf16_t* Vy = V + (size_t)y * NCOL * DD;

    // ---- Phase 1: 392 tiles (4 M-tiles x 98 N-tiles), 49 per wave.
    // tile = wid + 8*i: the 8 concurrent tiles cover all 4 M-tiles x 2 N-tiles
    // (A fragments shared by 2 waves, V fragments by 4 -> good L0 reuse).
    for (int i = 0; i < 49; ++i) {
        const int tile = wid + (i << 3);
        const int mt = tile & 3;
        const int nt = tile >> 2;
        const v16bf* Ap = (const v16bf*)(Ax + (size_t)(mt * 16 + lm) * DD + off);
        const v16bf* Bp = (const v16bf*)(Vy + (size_t)(nt * 16 + lm) * DD + off);

        // prefetch next N-tile's V fragment row toward L0 while WMMAs run
        if (i < 48) {
            __builtin_prefetch((const void*)(Vy + (size_t)((nt + 2) * 16 + lm) * DD + off), 0, 3);
        }

        v8f acc = {};
        #pragma unroll 4
        for (int k = 0; k < KSTEPS; ++k) {
            acc = __builtin_amdgcn_wmma_f32_16x16x32_bf16(
                false, Ap[2 * k], false, Bp[2 * k], (short)0, acc, false, false);
        }
        // C/D layout: lane holds column n0 = nt*16 + lane%16;
        // acc[r] is row mt*16 + r + 8*(lane>=16). Pad rows -> dump row (50).
        const int n0 = nt * 16 + lm;
        const int m0 = mt * 16 + ((lane & 16) ? 8 : 0);
        #pragma unroll
        for (int r = 0; r < 8; ++r) {
            const int row  = m0 + r;
            const int rowc = (row < NA) ? row : NA;   // branch-free clamp
            S[rowc * SSTR + n0] = acc[r];
        }
    }
    __syncthreads();

    // ---- Phase 2a: nonneg + temporal sums over the 50x1568 sims ----
    float nn = 0.0f, tp = 0.0f;
    for (int e = tid; e < NA * NCOL; e += 256) {
        const int a = e / NCOL;
        const int c = e - a * NCOL;
        const float s = S[a * SSTR + c] * invT;
        const float ng = fminf(fmaxf(s, -20.0f), 0.0f);
        nn += ng * ng;
        if (c >= NV) {
            const float d = s - S[a * SSTR + c - NV] * invT;
            tp += d * d;
        }
    }
    #pragma unroll
    for (int o = 16; o > 0; o >>= 1) {
        nn += __shfl_xor(nn, o, 32);
        tp += __shfl_xor(tp, o, 32);
    }
    if (lane == 0) { red[456 + wid] = nn; red[464 + wid] = tp; }

    // ---- Phase 2b: max over Nv for each (a,t) group ----
    for (int g = tid; g < NA * TT; g += 256) {
        const int a = g >> 3, t = g & 7;
        const float* rowp = S + a * SSTR + t * NV;
        float m = -1e30f;
        for (int i = 0; i < NV; ++i) m = fmaxf(m, rowp[i]);
        red[g] = m;
    }
    __syncthreads();

    if (tid < NA) {
        float cnt = 0.0f, ssum = 0.0f;
        #pragma unroll
        for (int t = 0; t < TT; ++t) {
            const float m = red[tid * 8 + t] * invT;
            if (m >= thr) { cnt += 1.0f; ssum += m; }
        }
        red[400 + tid] = ssum / fmaxf(cnt, 1.0f);
    }
    __syncthreads();

    if (tid == 0) {
        float s = 0.0f;
        for (int a = 0; a < NA; ++a) s += red[400 + a];
        clip[x * BSZ + y] = s * (1.0f / (float)NA);
        float a0 = 0.0f, a1 = 0.0f;
        for (int w = 0; w < 8; ++w) { a0 += red[456 + w]; a1 += red[464 + w]; }
        atomicAdd(&wsf[0], a0);
        atomicAdd(&wsf[1], a1);
    }
}

// ---------------------------------------------------------------------------
// Final scalar: InfoNCE on 24x24 clip_sims + regularizers.
// ---------------------------------------------------------------------------
__global__ void finalize_kernel(const float* __restrict__ wsf, const float* __restrict__ clip,
                                const float* __restrict__ tempP, float* __restrict__ out)
{
    if (threadIdx.x != 0) return;
    float contrast = 0.0f;
    for (int i = 0; i < BSZ; ++i) {
        float rm = -1e30f, cm = -1e30f;
        for (int j = 0; j < BSZ; ++j) {
            rm = fmaxf(rm, clip[i * BSZ + j]);
            cm = fmaxf(cm, clip[j * BSZ + i]);
        }
        float rs = 0.0f, cs = 0.0f;
        for (int j = 0; j < BSZ; ++j) {
            rs += expf(clip[i * BSZ + j] - rm);
            cs += expf(clip[j * BSZ + i] - cm);
        }
        const float diag = clip[i * BSZ + i];
        contrast += (logf(rs) + rm - diag) + (logf(cs) + cm - diag);
    }
    contrast = contrast / (float)BSZ * 0.5f;

    const float t  = tempP[0];
    const float lt = logf(t);
    float tl = fmaxf(logf(2.3f) - lt, 0.0f); tl = tl * tl * tl;
    float th = fmaxf(lt - logf(4.0f), 0.0f); th = th * th * th;

    const float l_nn = wsf[0] / ((float)BSZ * BSZ * NA * TT * NV);           // 45,158,400
    const float l_tp = wsf[1] / ((float)BSZ * BSZ * NA * (TT - 1) * NV);     // 39,513,600

    out[0] = contrast + 0.15f * l_nn + 8.0f * (tl + th) + 0.01f * l_tp;
}

// ---------------------------------------------------------------------------
extern "C" void kernel_launch(void* const* d_in, const int* in_sizes, int n_in,
                              void* d_out, int out_size, void* d_ws, size_t ws_size,
                              hipStream_t stream)
{
    const float* audio       = (const float*)d_in[0];  // (24,50,768)
    const float* visual      = (const float*)d_in[1];  // (24,8,196,768)
    const float* temperature = (const float*)d_in[2];  // scalar
    const float* threshold   = (const float*)d_in[3];  // scalar
    float* out = (float*)d_out;

    float*  wsf  = (float*)d_ws;                       // [0]=nonneg sum, [1]=temporal sum
    float*  clip = wsf + 16;                           // 576 floats
    bf16_t* wsA  = (bf16_t*)((char*)d_ws + 4096);                              // 24*64*768 bf16
    bf16_t* wsV  = (bf16_t*)((char*)d_ws + 4096 +
                             (size_t)BSZ * NA_PAD * DD * sizeof(bf16_t));      // 24*1568*768 bf16

    (void)in_sizes; (void)n_in; (void)out_size; (void)ws_size;

    hipFuncSetAttribute((const void*)gemm_pair_kernel,
                        hipFuncAttributeMaxDynamicSharedMemorySize, SMEM_BYTES);

    init_kernel<<<1, 32, 0, stream>>>(wsf);
    norm_audio_kernel<<<(BSZ * NA_PAD) / 8, 256, 0, stream>>>(audio, wsA);
    norm_visual_kernel<<<(BSZ * NCOL) / 8, 256, 0, stream>>>(visual, wsV);
    gemm_pair_kernel<<<dim3(BSZ, BSZ), 256, SMEM_BYTES, stream>>>(
        wsA, wsV, temperature, threshold, clip, wsf);
    finalize_kernel<<<1, 32, 0, stream>>>(wsf, clip, temperature, out);
}
